// GraphNeighborhoodReg_68848325755114
// MI455X (gfx1250) — matrix operand order repaired and checked by reading, not compiled, not run on hardware
//
#include <hip/hip_runtime.h>
#include <hip/hip_bf16.h>
#include <math.h>

// ---------------- problem constants ----------------
#define NN    8192
#define DD    64
#define KK    15
#define KP    16          // padded K stride
#define NNEG  32
#define TEMP_ 0.1f
#define MARGIN_ 0.5f
#define PPR_ALPHA_ 0.2f
#define PPR_ITERS_ 5
#define W_ALIGN_ 1.0f
#define W_CONTRAST_ 1.0f
#define W_LAPL_ 0.5f
#define W_DIFF_ 0.5f

typedef __attribute__((ext_vector_type(16))) _Float16 v16h;
typedef __attribute__((ext_vector_type(8)))  _Float16 v8h;
typedef __attribute__((ext_vector_type(8)))  float    v8f;

// ---------------- small utility kernels ----------------
__global__ void zero_acc_kernel(float* acc) {
    if (threadIdx.x < 16) acc[threadIdx.x] = 0.0f;
}

__global__ void zero_f4_kernel(float4* p, long n4) {
    long i = (long)blockIdx.x * blockDim.x + threadIdx.x;
    long stride = (long)gridDim.x * blockDim.x;
    float4 z; z.x = z.y = z.z = z.w = 0.0f;
    for (; i < n4; i += stride) p[i] = z;
}

__global__ void finalize_kernel(const float* acc, float* out) {
    if (threadIdx.x == 0 && blockIdx.x == 0) out[0] = acc[0];
}

// ---------------- normalize rows (f32 out + f16 out) ----------------
// one wave (32 lanes) per row of 64 floats
__global__ __launch_bounds__(256) void normalize_rows(const float* __restrict__ Z,
                                                      float* __restrict__ Zn,
                                                      _Float16* __restrict__ Zh) {
    int row  = blockIdx.x * 8 + (threadIdx.x >> 5);
    int lane = threadIdx.x & 31;
    float a = Z[(size_t)row * DD + lane];
    float b = Z[(size_t)row * DD + 32 + lane];
    float ss = a * a + b * b;
    #pragma unroll
    for (int o = 16; o > 0; o >>= 1) ss += __shfl_xor(ss, o, 32);
    float inv = 1.0f / fmaxf(sqrtf(ss), 1e-12f);
    float an = a * inv, bn = b * inv;
    Zn[(size_t)row * DD + lane] = an;
    Zn[(size_t)row * DD + 32 + lane] = bn;
    Zh[(size_t)row * DD + lane] = (_Float16)an;
    Zh[(size_t)row * DD + 32 + lane] = (_Float16)bn;
}

// ---------------- sim = Xh @ Xh^T via WMMA f32_16x16x32_f16 ----------------
// grid: (NN/16, NN/16/8), block 256 = 8 waves, one 16x16 tile per wave.
// A-frag (16x32 f16): lane L, half=L>>4, r=L&15, row m=m0+r,
//   v16h elems 0..7  = K [KB +      half*8, +8)   (contiguous 16B)
//   v16h elems 8..15 = K [KB + 16 + half*8, +8)   (contiguous 16B)
// B-frag (32x16 f16): col n=n0+r, same K layout as A-frag
__global__ __launch_bounds__(256) void sim_gemm(const _Float16* __restrict__ Xh,
                                                float* __restrict__ S) {
    const int lane  = threadIdx.x & 31;
    const int wave  = threadIdx.x >> 5;
    const int tileN = blockIdx.x;
    const int tileM = blockIdx.y * 8 + wave;
    const int half  = lane >> 4;
    const int r     = lane & 15;
    const int m = tileM * 16 + r;
    const int n = tileN * 16 + r;

    v16h a0, a1, b0, b1;
    {
        const v8h* pa00 = (const v8h*)(Xh + (size_t)m * DD + 0  + half * 8);
        const v8h* pa01 = (const v8h*)(Xh + (size_t)m * DD + 16 + half * 8);
        const v8h* pa10 = (const v8h*)(Xh + (size_t)m * DD + 32 + half * 8);
        const v8h* pa11 = (const v8h*)(Xh + (size_t)m * DD + 48 + half * 8);
        v8h x0 = *pa00, x1 = *pa01, x2 = *pa10, x3 = *pa11;
        const v8h* pb00 = (const v8h*)(Xh + (size_t)n * DD + 0  + half * 8);
        const v8h* pb01 = (const v8h*)(Xh + (size_t)n * DD + 16 + half * 8);
        const v8h* pb10 = (const v8h*)(Xh + (size_t)n * DD + 32 + half * 8);
        const v8h* pb11 = (const v8h*)(Xh + (size_t)n * DD + 48 + half * 8);
        v8h y0 = *pb00, y1 = *pb01, y2 = *pb10, y3 = *pb11;
        #pragma unroll
        for (int e = 0; e < 8; ++e) {
            a0[e] = x0[e]; a0[8 + e] = x1[e];
            a1[e] = x2[e]; a1[8 + e] = x3[e];
            b0[e] = y0[e]; b0[8 + e] = y1[e];
            b1[e] = y2[e]; b1[8 + e] = y3[e];
        }
    }
    v8f c = {};
    c = __builtin_amdgcn_wmma_f32_16x16x32_f16(false, a0, false, b0, (short)0, c, false, false);
    c = __builtin_amdgcn_wmma_f32_16x16x32_f16(false, a1, false, b1, (short)0, c, false, false);

    // C layout: VGPR q, lane L -> row m0 + q + 8*half, col n0 + (L&15)
    #pragma unroll
    for (int q = 0; q < 8; ++q) {
        int row = tileM * 16 + q + 8 * half;
        int col = tileN * 16 + r;
        S[(size_t)row * NN + col] = c[q];
    }
}

// ---------------- top-k(15) + softmax(temp) per row ----------------
__global__ __launch_bounds__(256) void topk_softmax(const float* __restrict__ S,
                                                    int* __restrict__ idx,
                                                    float* __restrict__ val) {
    __shared__ float row[NN];
    __shared__ float wv[8];
    __shared__ int   wi[8];
    __shared__ float selv[KP];
    __shared__ int   seli[KP];
    const int i = blockIdx.x;
    const float* srow = S + (size_t)i * NN;
    for (int c = threadIdx.x; c < NN; c += 256) row[c] = srow[c];
    __syncthreads();
    if (threadIdx.x == 0) row[i] = -INFINITY;  // mask self
    __syncthreads();

    for (int t = 0; t < KK; ++t) {
        float bv = -INFINITY; int bi = 0x7fffffff;
        for (int c = threadIdx.x; c < NN; c += 256) {
            float v = row[c];
            if (v > bv || (v == bv && c < bi)) { bv = v; bi = c; }
        }
        #pragma unroll
        for (int o = 16; o > 0; o >>= 1) {
            float ov = __shfl_xor(bv, o, 32);
            int   oi = __shfl_xor(bi, o, 32);
            if (ov > bv || (ov == bv && oi < bi)) { bv = ov; bi = oi; }
        }
        if ((threadIdx.x & 31) == 0) { wv[threadIdx.x >> 5] = bv; wi[threadIdx.x >> 5] = bi; }
        __syncthreads();
        if (threadIdx.x == 0) {
            float fv = wv[0]; int fi = wi[0];
            for (int q = 1; q < 8; ++q)
                if (wv[q] > fv || (wv[q] == fv && wi[q] < fi)) { fv = wv[q]; fi = wi[q]; }
            selv[t] = fv; seli[t] = fi;
            row[fi] = -INFINITY;
        }
        __syncthreads();
    }
    if (threadIdx.x == 0) {
        float mx = selv[0];
        float e[KK]; float s = 0.0f;
        #pragma unroll
        for (int k = 0; k < KK; ++k) { e[k] = expf((selv[k] - mx) / TEMP_); s += e[k]; }
        float is = 1.0f / s;
        #pragma unroll
        for (int k = 0; k < KK; ++k) {
            val[(size_t)i * KP + k] = e[k] * is;
            idx[(size_t)i * KP + k] = seli[k];
        }
        val[(size_t)i * KP + KK] = 0.0f;
        idx[(size_t)i * KP + KK] = -1;
    }
}

// ---------------- KL alignment (K-sparse x K-sparse) ----------------
__global__ void kl_kernel(const int* __restrict__ idxr, const float* __restrict__ valr,
                          const int* __restrict__ idxa, const float* __restrict__ vala,
                          float* acc) {
    int i = blockIdx.x * blockDim.x + threadIdx.x;
    if (i >= NN) return;
    float t = 0.0f;
    #pragma unroll
    for (int k = 0; k < KK; ++k) {
        int j = idxr[(size_t)i * KP + k];
        float p = valr[(size_t)i * KP + k];
        float q = 0.0f;
        #pragma unroll
        for (int m2 = 0; m2 < KK; ++m2)
            if (idxa[(size_t)i * KP + m2] == j) q = vala[(size_t)i * KP + m2];
        t += p * (logf(p) - logf(q + 1e-8f));
    }
    atomicAdd(acc, W_ALIGN_ * t / (float)NN);
}

// ---------------- contrastive (attract + hash-sampled negatives) ----------------
__global__ __launch_bounds__(256) void contrastive_kernel(const float* __restrict__ zan,
                                                          const int* __restrict__ idxr,
                                                          float* acc) {
    int i    = blockIdx.x * 8 + (threadIdx.x >> 5);
    int lane = threadIdx.x & 31;
    float zi0 = zan[(size_t)i * DD + lane];
    float zi1 = zan[(size_t)i * DD + 32 + lane];
    int nbr[KK];
    #pragma unroll
    for (int k = 0; k < KK; ++k) nbr[k] = idxr[(size_t)i * KP + k];

    float attr = 0.0f;
    #pragma unroll
    for (int k = 0; k < KK; ++k) {
        int j = nbr[k];
        float d = zi0 * zan[(size_t)j * DD + lane] + zi1 * zan[(size_t)j * DD + 32 + lane];
        #pragma unroll
        for (int o = 16; o > 0; o >>= 1) d += __shfl_xor(d, o, 32);
        attr += 1.0f - d;
    }

    float hinge = 0.0f;
    int accn = 0; unsigned ctr = 0;
    while (accn < NNEG && ctr < 512u) {
        unsigned h = (unsigned)i * 2654435761u ^ (ctr * 0x9E3779B9u);
        h ^= h >> 16; h *= 0x85EBCA6Bu; h ^= h >> 13; h *= 0xC2B2AE35u; h ^= h >> 16;
        int j = (int)(h & (NN - 1));
        ++ctr;
        bool bad = (j == i);
        #pragma unroll
        for (int k = 0; k < KK; ++k) bad = bad || (nbr[k] == j);
        if (bad) continue;
        float d = zi0 * zan[(size_t)j * DD + lane] + zi1 * zan[(size_t)j * DD + 32 + lane];
        #pragma unroll
        for (int o = 16; o > 0; o >>= 1) d += __shfl_xor(d, o, 32);
        float cd = 1.0f - d;
        hinge += fmaxf(MARGIN_ - cd, 0.0f);
        ++accn;
    }
    if (lane == 0)
        atomicAdd(acc, W_CONTRAST_ * (attr / ((float)KK * NN) + hinge / ((float)NN * NNEG)));
}

// ---------------- Laplacian (row degree == 15 exactly) ----------------
__global__ __launch_bounds__(256) void lapl_kernel(const float* __restrict__ za,
                                                   const int* __restrict__ idxr,
                                                   float* acc) {
    int i    = blockIdx.x * 8 + (threadIdx.x >> 5);
    int lane = threadIdx.x & 31;
    float zi0 = za[(size_t)i * DD + lane];
    float zi1 = za[(size_t)i * DD + 32 + lane];
    float dii = zi0 * zi0 + zi1 * zi1;
    #pragma unroll
    for (int o = 16; o > 0; o >>= 1) dii += __shfl_xor(dii, o, 32);
    float s = 0.0f;
    #pragma unroll
    for (int k = 0; k < KK; ++k) {
        int j = idxr[(size_t)i * KP + k];
        float d = zi0 * za[(size_t)j * DD + lane] + zi1 * za[(size_t)j * DD + 32 + lane];
        #pragma unroll
        for (int o = 16; o > 0; o >>= 1) d += __shfl_xor(d, o, 32);
        s += d;
    }
    if (lane == 0)
        atomicAdd(acc, W_LAPL_ * (dii - s / (float)KK) / (float)NN);
}

// ---------------- PPR: S0 scatter, sparse-row iterate, fixup ----------------
__global__ void scatter_s0(float* __restrict__ S, const int* __restrict__ idx,
                           const float* __restrict__ val) {
    int t = blockIdx.x * blockDim.x + threadIdx.x;
    int i = t >> 4, k = t & 15;
    if (i < NN && k < KK)
        S[(size_t)i * NN + idx[(size_t)i * KP + k]] = val[(size_t)i * KP + k];
}

__global__ __launch_bounds__(256) void ppr_iter(const float* __restrict__ Sin,
                                                float* __restrict__ Sout,
                                                const int* __restrict__ idx,
                                                const float* __restrict__ val) {
    int i = blockIdx.x;
    __shared__ int   ji[KK];
    __shared__ float jw[KK];
    if (threadIdx.x < KK) {
        ji[threadIdx.x] = idx[(size_t)i * KP + threadIdx.x];
        jw[threadIdx.x] = val[(size_t)i * KP + threadIdx.x] * (1.0f - PPR_ALPHA_);
    }
    __syncthreads();
    const float4* S4 = (const float4*)Sin;
    float4* O4 = (float4*)(Sout + (size_t)i * NN);
    for (int c = threadIdx.x; c < NN / 4; c += 256) {
        float4 a; a.x = a.y = a.z = a.w = 0.0f;
        #pragma unroll
        for (int k = 0; k < KK; ++k) {
            const float4* p = &S4[(size_t)ji[k] * (NN / 4) + c];
            __builtin_prefetch(p + 256, 0, 1);   // global_prefetch next chunk
            float4 v = *p;
            float  w = jw[k];
            a.x += w * v.x; a.y += w * v.y; a.z += w * v.z; a.w += w * v.w;
        }
        O4[c] = a;
    }
}

__global__ void ppr_fixup(float* __restrict__ S, const int* __restrict__ idx,
                          const float* __restrict__ val) {
    int t = blockIdx.x * blockDim.x + threadIdx.x;
    int i = t >> 4, k = t & 15;
    if (i < NN && k < KK)
        S[(size_t)i * NN + idx[(size_t)i * KP + k]] += PPR_ALPHA_ * val[(size_t)i * KP + k];
}

// ---------------- diffusion loss: mean((A_atac - S/rowsum)^2) ----------------
__global__ __launch_bounds__(256) void diff_kernel(const float* __restrict__ S,
                                                   const int* __restrict__ idxa,
                                                   const float* __restrict__ vala,
                                                   float* acc) {
    __shared__ float wr[8], wq[8];
    int i = blockIdx.x;
    const float* row = S + (size_t)i * NN;
    float rs = 0.0f, sq = 0.0f;
    for (int c = threadIdx.x; c < NN; c += 256) {
        float v = row[c];
        rs += v; sq += v * v;
    }
    #pragma unroll
    for (int o = 16; o > 0; o >>= 1) { rs += __shfl_xor(rs, o, 32); sq += __shfl_xor(sq, o, 32); }
    if ((threadIdx.x & 31) == 0) { wr[threadIdx.x >> 5] = rs; wq[threadIdx.x >> 5] = sq; }
    __syncthreads();
    if (threadIdx.x == 0) {
        float R = 0.0f, Q = 0.0f;
        for (int q = 0; q < 8; ++q) { R += wr[q]; Q += wq[q]; }
        R = fmaxf(R, 1e-8f);
        float cross = 0.0f, selfsq = 0.0f;
        #pragma unroll
        for (int k = 0; k < KK; ++k) {
            int j = idxa[(size_t)i * KP + k];
            float v = vala[(size_t)i * KP + k];
            cross += v * row[j] / R;
            selfsq += v * v;
        }
        float contrib = Q / (R * R) - 2.0f * cross + selfsq;
        atomicAdd(acc, W_DIFF_ * contrib / ((float)NN * (float)NN));
    }
}

// ---------------- driver ----------------
extern "C" void kernel_launch(void* const* d_in, const int* in_sizes, int n_in,
                              void* d_out, int out_size, void* d_ws, size_t ws_size,
                              hipStream_t stream) {
    const float* z_rna  = (const float*)d_in[0];
    const float* z_atac = (const float*)d_in[1];
    float* out = (float*)d_out;

    const size_t bufBytes = (size_t)NN * NN * sizeof(float);   // 256 MB each
    char* p = (char*)d_ws;
    float*    buf0 = (float*)p;            p += bufBytes;      // sim_rna -> S ping
    float*    buf1 = (float*)p;            p += bufBytes;      // sim_atac -> S pong
    float*    zrn  = (float*)p;            p += (size_t)NN * DD * sizeof(float);
    float*    zan  = (float*)p;            p += (size_t)NN * DD * sizeof(float);
    _Float16* zrh  = (_Float16*)p;         p += (size_t)NN * DD * sizeof(_Float16);
    _Float16* zah  = (_Float16*)p;         p += (size_t)NN * DD * sizeof(_Float16);
    int*      idxr = (int*)p;              p += (size_t)NN * KP * sizeof(int);
    float*    valr = (float*)p;            p += (size_t)NN * KP * sizeof(float);
    int*      idxa = (int*)p;              p += (size_t)NN * KP * sizeof(int);
    float*    vala = (float*)p;            p += (size_t)NN * KP * sizeof(float);
    float*    acc  = (float*)p;            p += 256;

    zero_acc_kernel<<<1, 64, 0, stream>>>(acc);

    // normalize both modalities (f32 + f16 copies)
    normalize_rows<<<NN / 8, 256, 0, stream>>>(z_rna,  zrn, zrh);
    normalize_rows<<<NN / 8, 256, 0, stream>>>(z_atac, zan, zah);

    // sim matrices via WMMA
    dim3 ggrid(NN / 16, NN / 16 / 8);
    sim_gemm<<<ggrid, 256, 0, stream>>>(zrh, buf0);
    topk_softmax<<<NN, 256, 0, stream>>>(buf0, idxr, valr);
    sim_gemm<<<ggrid, 256, 0, stream>>>(zah, buf1);
    topk_softmax<<<NN, 256, 0, stream>>>(buf1, idxa, vala);

    // K-sparse loss terms
    kl_kernel<<<NN / 256, 256, 0, stream>>>(idxr, valr, idxa, vala, acc);
    contrastive_kernel<<<NN / 8, 256, 0, stream>>>(zan, idxr, acc);
    lapl_kernel<<<NN / 8, 256, 0, stream>>>(z_atac, idxr, acc);

    // PPR diffusion: S0 = A_rna_soft (K-sparse in dense buffer), 5 sparse-gather iters
    zero_f4_kernel<<<4096, 256, 0, stream>>>((float4*)buf0, (long)NN * NN / 4);
    scatter_s0<<<NN * KP / 256, 256, 0, stream>>>(buf0, idxr, valr);
    float* cur = buf0;
    float* nxt = buf1;
    for (int it = 0; it < PPR_ITERS_; ++it) {
        ppr_iter<<<NN, 256, 0, stream>>>(cur, nxt, idxr, valr);
        ppr_fixup<<<NN * KP / 256, 256, 0, stream>>>(nxt, idxr, valr);
        float* t = cur; cur = nxt; nxt = t;
    }
    // final S is in `cur` (buf1 after 5 iterations)
    diff_kernel<<<NN, 256, 0, stream>>>(cur, idxa, vala, acc);

    finalize_kernel<<<1, 32, 0, stream>>>(acc, out);
    (void)in_sizes; (void)n_in; (void)out_size; (void)ws_size;
}
